// TripletInBatchNegSoftmaxContrastLoss_22497038696605
// MI455X (gfx1250) — compile-verified, matrix operand order checked by reading, not run on hardware
//
#include <hip/hip_runtime.h>

typedef __attribute__((ext_vector_type(16))) _Float16     v16h;
typedef __attribute__((ext_vector_type(8)))  _Float16     v8h;
typedef __attribute__((ext_vector_type(8)))  float        v8f;
typedef __attribute__((ext_vector_type(4)))  unsigned int v4u;
typedef __attribute__((ext_vector_type(4)))  int          v4i;
typedef __attribute__((ext_vector_type(8)))  int          v8i;

#define B_ROWS  8192
#define DIM     1024
#define INV_T   20.0f         // 1 / 0.05
#define NCOLBLK 64            // 8192 / 128 column blocks
#define KSTAGE  64            // K elements per TDM stage
#define LDS_STRIDE 72         // 64 halves + 8 halves TDM pad = 144 B (bank-conflict-free)

// ---------------------------------------------------------------------------
// Kernel 0: zero the scalar output (harness poisons d_out with 0xAA).
// ---------------------------------------------------------------------------
__global__ void zero_kernel(float* out) {
    if (threadIdx.x == 0 && blockIdx.x == 0) out[0] = 0.0f;
}

// ---------------------------------------------------------------------------
// Kernel 1: per-row L2 norms (eps-clamped), write normalized a,n as f16,
// and the diagonal positive cosine similarity. One block (256 thr) per row.
// ---------------------------------------------------------------------------
__global__ __launch_bounds__(256)
void normalize_kernel(const float* __restrict__ a,
                      const float* __restrict__ p,
                      const float* __restrict__ n,
                      _Float16* __restrict__ a16,
                      _Float16* __restrict__ n16,
                      float* __restrict__ pos)
{
    const int row = blockIdx.x;
    const int tid = threadIdx.x;
    const float* ar = a + (size_t)row * DIM;
    const float* pr = p + (size_t)row * DIM;
    const float* nr = n + (size_t)row * DIM;

    float sa = 0.f, sp = 0.f, sn = 0.f, dap = 0.f;
    float av[4], nv[4];
#pragma unroll
    for (int i = 0; i < 4; ++i) {
        const int j = tid + i * 256;
        const float x = ar[j], y = pr[j], z = nr[j];
        av[i] = x; nv[i] = z;
        sa += x * x; sp += y * y; sn += z * z; dap += x * y;
    }
#pragma unroll
    for (int off = 16; off; off >>= 1) {
        sa  += __shfl_xor(sa,  off, 32);
        sp  += __shfl_xor(sp,  off, 32);
        sn  += __shfl_xor(sn,  off, 32);
        dap += __shfl_xor(dap, off, 32);
    }
    __shared__ float red[8][4];
    __shared__ float bc[2];
    const int wave = tid >> 5, lane = tid & 31;
    if (lane == 0) { red[wave][0] = sa; red[wave][1] = sp; red[wave][2] = sn; red[wave][3] = dap; }
    __syncthreads();
    if (tid == 0) {
        float ta = 0.f, tp = 0.f, tn = 0.f, td = 0.f;
#pragma unroll
        for (int w = 0; w < 8; ++w) { ta += red[w][0]; tp += red[w][1]; tn += red[w][2]; td += red[w][3]; }
        const float na = fmaxf(sqrtf(ta), 1e-8f);
        const float np = fmaxf(sqrtf(tp), 1e-8f);
        const float nn = fmaxf(sqrtf(tn), 1e-8f);
        bc[0] = 1.0f / na;
        bc[1] = 1.0f / nn;
        pos[row] = td / (na * np);
    }
    __syncthreads();
    const float inva = bc[0], invn = bc[1];
#pragma unroll
    for (int i = 0; i < 4; ++i) {
        const int j = tid + i * 256;
        a16[(size_t)row * DIM + j] = (_Float16)(av[i] * inva);
        n16[(size_t)row * DIM + j] = (_Float16)(nv[i] * invn);
    }
}

// ---------------------------------------------------------------------------
// TDM: issue one tensor_load_to_lds of a 128(rows) x 64(K halves) tile.
//   data_size=2B, tile_dim0=64 (contig K), tile_dim1=128 rows,
//   tensor_dim0_stride=1024, pad: +4 dwords after every 32 dwords stored
//   -> LDS row stride 144 B (= LDS_STRIDE halves), bank-conflict-free.
// ---------------------------------------------------------------------------
__device__ __forceinline__ void tdm_load_tile_128x64(const _Float16* gtile,
                                                     unsigned ldsByteAddr)
{
    const unsigned long long ga = (unsigned long long)(size_t)gtile;
    v4u g0;
    g0.x = 1u;                                        // count=1 (valid user D#)
    g0.y = ldsByteAddr;                               // lds_addr (bytes)
    g0.z = (unsigned)(ga & 0xffffffffu);              // global_addr[31:0]
    g0.w = (unsigned)((ga >> 32) & 0x01ffffffu)       // global_addr[56:32]
         | 0x80000000u;                               // type=2 ("image")
    v8i g1;
    g1[0] = 0x07110000;         // data_size=1(2B) | pad_en | pad_int=4(32dw) | pad_amt=3(4dw)
    g1[1] = (int)(0x0400u << 16);   // tensor_dim0 = 1024 (low16 in [31:16])
    g1[2] = (int)(0x2000u << 16);   // tensor_dim0 hi=0 | tensor_dim1 = 8192 low16
    g1[3] = (int)(64u << 16);       // tensor_dim1 hi=0 | tile_dim0 = 64
    g1[4] = 128;                    // tile_dim1 = 128 | tile_dim2 = 0 (2D)
    g1[5] = 1024;                   // tensor_dim0_stride low32 (units of 2B)
    g1[6] = 0;                      // stride hi | tensor_dim1_stride low (unused, 2D)
    g1[7] = 0;
    const v4i gz = {0, 0, 0, 0};
#if __clang_major__ >= 23
    const v8i gz8 = {0, 0, 0, 0, 0, 0, 0, 0};
    __builtin_amdgcn_tensor_load_to_lds(g0, g1, gz, gz, gz8, 0);
#else
    __builtin_amdgcn_tensor_load_to_lds(g0, g1, gz, gz, 0);
#endif
}

// ---------------------------------------------------------------------------
// Kernel 2: fused GEMM + partial logsumexp, TDM-fed double-buffered LDS.
// Grid (64,64): block (x,y) computes rows [128x,+128) x cols [128y,+128).
// 8 wave32 as 4(M) x 2(N); each wave owns 32x64 = 2x4 wmma f16 accumulators.
// Wave 0 drives the Tensor Data Mover (2 tiles/stage, 2 stages in flight).
// ---------------------------------------------------------------------------
__global__ __launch_bounds__(256)
void gemm_lse_kernel(const _Float16* __restrict__ a16,
                     const _Float16* __restrict__ n16,
                     float* __restrict__ pm,
                     float* __restrict__ pl)
{
    __shared__ _Float16 smA[2][128 * LDS_STRIDE];   // 2 x 18 KB
    __shared__ _Float16 smB[2][128 * LDS_STRIDE];   // 2 x 18 KB
    __shared__ float smM[128][2];
    __shared__ float smS[128][2];

    const int rowBase = blockIdx.x * 128;
    const int colBase = blockIdx.y * 128;
    const int tid    = threadIdx.x;
    const int wave   = tid >> 5;
    const int lane   = tid & 31;
    const int mwave  = wave & 3;           // rows  mwave*32 .. +32
    const int nwave  = wave >> 2;          // cols  nwave*64 .. +64
    const int lane15 = lane & 15;
    const int khalf  = (lane >> 4) << 3;   // 0 or 8 (16-bit A/B K interleave)

    const _Float16* gA = a16 + (size_t)rowBase * DIM;   // A tile column origin
    const _Float16* gB = n16 + (size_t)colBase * DIM;   // B tile column origin

    v8f acc[2][4] = {};

    constexpr int NSTAGE = DIM / KSTAGE;    // 16
    // Prologue: stage 0 in flight.
    if (wave == 0) {
        tdm_load_tile_128x64(gA, (unsigned)(size_t)(void*)&smA[0][0]);
        tdm_load_tile_128x64(gB, (unsigned)(size_t)(void*)&smB[0][0]);
    }

    for (int s = 0; s < NSTAGE; ++s) {
        const int buf = s & 1;
        if (wave == 0) {
            if (s + 1 < NSTAGE) {   // issue next stage, then wait for current
                const int nb = (s + 1) & 1;
                tdm_load_tile_128x64(gA + (s + 1) * KSTAGE,
                                     (unsigned)(size_t)(void*)&smA[nb][0]);
                tdm_load_tile_128x64(gB + (s + 1) * KSTAGE,
                                     (unsigned)(size_t)(void*)&smB[nb][0]);
                __builtin_amdgcn_s_wait_tensorcnt(2);   // stage s complete
            } else {
                __builtin_amdgcn_s_wait_tensorcnt(0);
            }
        }
        __syncthreads();            // stage-s tiles visible to all waves

        const _Float16* Ab = &smA[buf][0];
        const _Float16* Bb = &smB[buf][0];
#pragma unroll
        for (int ks = 0; ks < KSTAGE / 32; ++ks) {      // 2 x K=32 sub-steps
            v16h af[2], bf[4];
#pragma unroll
            for (int m = 0; m < 2; ++m) {
                const int r = mwave * 32 + m * 16 + lane15;
                const _Float16* pA = Ab + r * LDS_STRIDE + ks * 32 + khalf;
                v8h lo = *(const v8h*)(pA);         // K +0..7   (lane half)
                v8h hi = *(const v8h*)(pA + 16);    // K +16..23 (lane half)
                af[m] = __builtin_shufflevector(lo, hi, 0,1,2,3,4,5,6,7,8,9,10,11,12,13,14,15);
            }
#pragma unroll
            for (int nn = 0; nn < 4; ++nn) {
                const int c = nwave * 64 + nn * 16 + lane15;
                const _Float16* pB = Bb + c * LDS_STRIDE + ks * 32 + khalf;
                v8h lo = *(const v8h*)(pB);
                v8h hi = *(const v8h*)(pB + 16);
                bf[nn] = __builtin_shufflevector(lo, hi, 0,1,2,3,4,5,6,7,8,9,10,11,12,13,14,15);
            }
#pragma unroll
            for (int m = 0; m < 2; ++m)
#pragma unroll
                for (int nn = 0; nn < 4; ++nn)
                    acc[m][nn] = __builtin_amdgcn_wmma_f32_16x16x32_f16(
                        false, af[m], false, bf[nn], (short)0, acc[m][nn], false, false);
        }
        __syncthreads();            // done reading buf before TDM reuses it
    }

    // ---- fused partial logsumexp epilogue --------------------------------
    // C layout: VGPR r, lanes 0-15 -> (row r, N=lane), lanes 16-31 -> row r+8.
#pragma unroll
    for (int m = 0; m < 2; ++m)
#pragma unroll
        for (int r = 0; r < 8; ++r) {
            float t = fmaxf(fmaxf(acc[m][0][r], acc[m][1][r]),
                            fmaxf(acc[m][2][r], acc[m][3][r]));
            t = fmaxf(t, __shfl_xor(t, 1, 32));
            t = fmaxf(t, __shfl_xor(t, 2, 32));
            t = fmaxf(t, __shfl_xor(t, 4, 32));
            t = fmaxf(t, __shfl_xor(t, 8, 32));
            if (lane15 == 0) {
                const int rowLoc = mwave * 32 + m * 16 + r + ((lane >> 4) << 3);
                smM[rowLoc][nwave] = t * INV_T;
            }
        }
    __syncthreads();
#pragma unroll
    for (int m = 0; m < 2; ++m)
#pragma unroll
        for (int r = 0; r < 8; ++r) {
            const int rowLoc = mwave * 32 + m * 16 + r + ((lane >> 4) << 3);
            const float rm = fmaxf(smM[rowLoc][0], smM[rowLoc][1]);
            float s = __expf(acc[m][0][r] * INV_T - rm)
                    + __expf(acc[m][1][r] * INV_T - rm)
                    + __expf(acc[m][2][r] * INV_T - rm)
                    + __expf(acc[m][3][r] * INV_T - rm);
            s += __shfl_xor(s, 1, 32);
            s += __shfl_xor(s, 2, 32);
            s += __shfl_xor(s, 4, 32);
            s += __shfl_xor(s, 8, 32);
            if (lane15 == 0) smS[rowLoc][nwave] = s;
        }
    __syncthreads();

    if (tid < 128) {
        const int row = tid;
        const float M = fmaxf(smM[row][0], smM[row][1]);
        const float L = smS[row][0] + smS[row][1];
        pm[(size_t)(rowBase + row) * NCOLBLK + blockIdx.y] = M;
        pl[(size_t)(rowBase + row) * NCOLBLK + blockIdx.y] = L;
    }
}

// ---------------------------------------------------------------------------
// Kernel 3: merge 64 partial logsumexps per row + positive logit, accumulate
// mean(lse - logit0) into out. One 64-thread block (2 waves) per row.
// ---------------------------------------------------------------------------
__global__ __launch_bounds__(64)
void finalize_kernel(const float* __restrict__ pm,
                     const float* __restrict__ pl,
                     const float* __restrict__ pos,
                     float* __restrict__ out)
{
    const int row = blockIdx.x;
    const int t = threadIdx.x;
    const float m = pm[(size_t)row * NCOLBLK + t];
    const float l = pl[(size_t)row * NCOLBLK + t];

    float mm = m;
#pragma unroll
    for (int off = 16; off; off >>= 1) mm = fmaxf(mm, __shfl_xor(mm, off, 32));
    __shared__ float sM[2], sS[2];
    const int wave = t >> 5, lane = t & 31;
    if (lane == 0) sM[wave] = mm;
    __syncthreads();
    const float M = fmaxf(sM[0], sM[1]);
    float e = l * __expf(m - M);
#pragma unroll
    for (int off = 16; off; off >>= 1) e += __shfl_xor(e, off, 32);
    if (lane == 0) sS[wave] = e;
    __syncthreads();
    if (t == 0) {
        const float S  = sS[0] + sS[1];
        const float z  = pos[row] * INV_T;           // positive logit (column 0)
        const float Mf = fmaxf(M, z);
        const float lse = Mf + __logf(S * __expf(M - Mf) + __expf(z - Mf));
        atomicAdd(out, (lse - z) * (1.0f / (float)B_ROWS));
    }
}

// ---------------------------------------------------------------------------
extern "C" void kernel_launch(void* const* d_in, const int* in_sizes, int n_in,
                              void* d_out, int out_size, void* d_ws, size_t ws_size,
                              hipStream_t stream)
{
    const float* a = (const float*)d_in[0];   // text_embeddings      [8192,1024] f32
    const float* p = (const float*)d_in[1];   // text_pos_embeddings
    const float* n = (const float*)d_in[2];   // text_neg_embeddings
    float* out = (float*)d_out;

    char* ws = (char*)d_ws;
    _Float16* a16 = (_Float16*)ws;                                     // 16 MB
    _Float16* n16 = (_Float16*)(ws + (size_t)B_ROWS * DIM * 2);        // 16 MB
    float* pos = (float*)(ws + (size_t)B_ROWS * DIM * 4);              // 32 KB
    float* pm  = pos + B_ROWS;                                         // 2 MB
    float* pl  = pm + (size_t)B_ROWS * NCOLBLK;                        // 2 MB

    zero_kernel<<<1, 64, 0, stream>>>(out);
    normalize_kernel<<<B_ROWS, 256, 0, stream>>>(a, p, n, a16, n16, pos);
    gemm_lse_kernel<<<dim3(B_ROWS / 128, B_ROWS / 128), 256, 0, stream>>>(a16, n16, pm, pl);
    finalize_kernel<<<B_ROWS, 64, 0, stream>>>(pm, pl, pos, out);
}
